// BertSelfAttention_14757507629733
// MI455X (gfx1250) — matrix operand order repaired and checked by reading, not compiled, not run on hardware
//
#include <hip/hip_runtime.h>
#include <hip/hip_bf16.h>
#include <math.h>

typedef __attribute__((ext_vector_type(16))) _Float16 v16h;
typedef __attribute__((ext_vector_type(8)))  _Float16 v8h;
typedef __attribute__((ext_vector_type(8)))  float    v8f;
typedef __attribute__((ext_vector_type(4)))  unsigned int u32x4;
typedef __attribute__((ext_vector_type(8)))  int          i32x8;
typedef __attribute__((ext_vector_type(4)))  int          i32x4;

#define HIDDEN 768
#define NHEADS 12
#define HDIM   64
#define BATCH  4
#define SEQ    2048
#define ROWS   (BATCH*SEQ)   // 8192

#define HAVE_TDM __has_builtin(__builtin_amdgcn_tensor_load_to_lds)

// ---------------- conversion kernels ----------------
__global__ void k_cvt_f16(const float* __restrict__ in, _Float16* __restrict__ out, int n) {
  int i = blockIdx.x * 256 + threadIdx.x;
  if (i < n) out[i] = (_Float16)in[i];
}

// W[k][n] (768x768 f32) -> Wt[n][k] (f16)
__global__ void k_transposeW(const float* __restrict__ in, _Float16* __restrict__ out) {
  __shared__ float tile[32][33];
  int bx = blockIdx.x * 32, by = blockIdx.y * 32;
  int tx = threadIdx.x, ty = threadIdx.y;
#pragma unroll
  for (int i = ty; i < 32; i += 8) tile[i][tx] = in[(size_t)(by + i) * HIDDEN + bx + tx];
  __syncthreads();
#pragma unroll
  for (int i = ty; i < 32; i += 8) out[(size_t)(bx + i) * HIDDEN + by + tx] = (_Float16)tile[tx][i];
}

// ---------------- WMMA GEMM:  C[m][n] = sum_k A[m][k]*Bt[n][k] + bias[n] ----------------
// MODE 0: f16 row-major out   (Q, K projections)
// MODE 1: f16 transposed out  (V projection -> Vt[bh][d][s])
// MODE 2: f32 out with exact erf-GELU (output projection -> d_out)
#define BM 128
#define BN 128
#define BK 32
#define LDT 40   // padded LDS row stride (halfs): 32 data + 8 pad = TDM 16DW interval + 4DW pad

#if HAVE_TDM
// Issue one TDM 2D tile load: 128 rows x 32 halfs, row stride 768 halfs, into LDS
// with per-row padding so rows land at stride LDT(=40) halfs.
__device__ __forceinline__ void tdm_tile_2d(const _Float16* gsrc, unsigned lds_byte) {
  unsigned long long ga = (unsigned long long)(uintptr_t)gsrc;
  u32x4 g0;
  g0[0] = 1u;                                       // count=1, user mode
  g0[1] = lds_byte;                                 // lds_addr
  g0[2] = (unsigned)(ga & 0xffffffffu);             // global_addr[31:0]
  g0[3] = (unsigned)((ga >> 32) & 0x01ffffffu)      // global_addr[56:32]
        | (2u << 30);                               // type = 2 (image/tensor)
  i32x8 g1;
  g1[0] = (1 << 16)                                 // data_size = 2 bytes
        | (1 << 20)                                 // pad_enable
        | (3 << 22)                                 // pad_interval: 16 DWORDs (=32 halfs)
        | (3 << 25);                                // pad_amount:   4 DWORDs (=8 halfs)
  g1[1] = (HIDDEN & 0xffff) << 16;                  // tensor_dim0 = 768 (low bits)
  g1[2] = (BM & 0xffff) << 16;                      // tensor_dim1 = 128 rows
  g1[3] = (BK << 16);                               // tile_dim0 = 32 (dim1 hi = 0)
  g1[4] = BM;                                       // tile_dim1 = 128, tile_dim2 = 0
  g1[5] = HIDDEN;                                   // tensor_dim0_stride = 768 (low 32)
  g1[6] = 0;                                        // stride hi, dim1_stride lo
  g1[7] = 0;
  i32x4 gz4 = {0, 0, 0, 0};                         // groups 2/3 unused (2D)
  i32x8 gz8 = {0, 0, 0, 0, 0, 0, 0, 0};             // extra group (6-arg form), unused
  __builtin_amdgcn_tensor_load_to_lds(g0, g1, gz4, gz4, gz8, 0);
}
#endif

template <int MODE>
__global__ __launch_bounds__(256)
void k_gemm(const _Float16* __restrict__ A, const _Float16* __restrict__ Bt,
            const float* __restrict__ bias, _Float16* __restrict__ outh,
            float* __restrict__ outf) {
  __shared__ _Float16 As[2][BM * LDT];
  __shared__ _Float16 Bs[2][BN * LDT];

  int tid = threadIdx.x;
  int wave = tid >> 5, lane = tid & 31;
  int lm = lane & 15, kh = lane >> 4;
  int wm = wave & 3, wn = wave >> 2;      // 4x2 waves; wave tile = 32x64
  int mbase = blockIdx.x * BM;
  int nbase = blockIdx.y * BN;

  v8f zero = {};
  v8f acc[2][4];
#pragma unroll
  for (int i = 0; i < 2; i++)
#pragma unroll
    for (int j = 0; j < 4; j++) acc[i][j] = zero;

  const _Float16* Abase = A  + (size_t)mbase * HIDDEN;
  const _Float16* Bbase = Bt + (size_t)nbase * HIDDEN;

  // stage one (A,B) tile pair for k-offset kb into buffer `buf`
  auto stage = [&](int buf, int kb) {
#if HAVE_TDM
    if (wave == 0) {
      tdm_tile_2d(Abase + kb, (unsigned)(unsigned long long)(uintptr_t)&As[buf][0]);
      tdm_tile_2d(Bbase + kb, (unsigned)(unsigned long long)(uintptr_t)&Bs[buf][0]);
    }
#else
    int lrow = tid >> 1;               // 0..127
    int lcol = (tid & 1) * 16;         // 0 or 16
    const _Float16* ga = Abase + (size_t)lrow * HIDDEN + kb + lcol;
    const _Float16* gb = Bbase + (size_t)lrow * HIDDEN + kb + lcol;
    *(v8h*)&As[buf][lrow * LDT + lcol]     = *(const v8h*)(ga);
    *(v8h*)&As[buf][lrow * LDT + lcol + 8] = *(const v8h*)(ga + 8);
    *(v8h*)&Bs[buf][lrow * LDT + lcol]     = *(const v8h*)(gb);
    *(v8h*)&Bs[buf][lrow * LDT + lcol + 8] = *(const v8h*)(gb + 8);
#endif
  };

  stage(0, 0);
  int buf = 0;
  for (int kb = 0; kb < HIDDEN; kb += BK) {
    int nkb = kb + BK;
    if (nkb < HIDDEN) {
      stage(buf ^ 1, nkb);
#if HAVE_TDM
      if (wave == 0) __builtin_amdgcn_s_wait_tensorcnt(2);  // current pair retired
#endif
    } else {
#if HAVE_TDM
      if (wave == 0) __builtin_amdgcn_s_wait_tensorcnt(0);
#endif
    }
    __syncthreads();

    v16h af[2], bf[4];
#pragma unroll
    for (int i = 0; i < 2; i++) {
      const _Float16* p = &As[buf][(wm * 32 + i * 16 + lm) * LDT];
      ((v8h*)&af[i])[0] = *(const v8h*)(p + kh * 8);
      ((v8h*)&af[i])[1] = *(const v8h*)(p + 16 + kh * 8);
    }
#pragma unroll
    for (int j = 0; j < 4; j++) {
      const _Float16* p = &Bs[buf][(wn * 64 + j * 16 + lm) * LDT];
      ((v8h*)&bf[j])[0] = *(const v8h*)(p + kh * 8);
      ((v8h*)&bf[j])[1] = *(const v8h*)(p + 16 + kh * 8);
    }
#pragma unroll
    for (int i = 0; i < 2; i++)
#pragma unroll
      for (int j = 0; j < 4; j++)
        acc[i][j] = __builtin_amdgcn_wmma_f32_16x16x32_f16(
            false, af[i], false, bf[j], (short)0, acc[i][j], false, false);
    __syncthreads();
    buf ^= 1;
  }

#pragma unroll
  for (int i = 0; i < 2; i++) {
#pragma unroll
    for (int j = 0; j < 4; j++) {
      int n = nbase + wn * 64 + j * 16 + lm;
      float bv = bias[n];
#pragma unroll
      for (int r = 0; r < 8; r++) {
        int m = mbase + wm * 32 + i * 16 + kh * 8 + r;
        float v = acc[i][j][r] + bv;
        if (MODE == 0) {
          outh[(size_t)m * HIDDEN + n] = (_Float16)v;
        } else if (MODE == 1) {
          int b = m >> 11, s = m & (SEQ - 1);
          int h = n >> 6, d = n & 63;
          outh[((size_t)(b * NHEADS + h) * HDIM + d) * SEQ + s] = (_Float16)v;
        } else {
          outf[(size_t)m * HIDDEN + n] = 0.5f * v * (1.0f + erff(v * 0.70710678118654752f));
        }
      }
    }
  }
}

// ---------------- flash attention: one wave per (bh, 16-row q tile) ----------------
__global__ __launch_bounds__(32)
void k_attn(const _Float16* __restrict__ Qh, const _Float16* __restrict__ Kh,
            const _Float16* __restrict__ Vt, const float* __restrict__ mask,
            _Float16* __restrict__ Ctx) {
  __shared__ _Float16 Pl[16 * 40];
  int bh = blockIdx.y;
  int b = bh / NHEADS, h = bh % NHEADS;
  int qbase = blockIdx.x * 16;
  int lane = threadIdx.x, lm = lane & 15, kh = lane >> 4;

  // Q fragments (A layout), 2 chunks covering d=0..63
  const _Float16* qrow = Qh + (size_t)(b * SEQ + qbase + lm) * HIDDEN + h * HDIM;
  v16h qa[2];
#pragma unroll
  for (int c = 0; c < 2; c++) {
    ((v8h*)&qa[c])[0] = *(const v8h*)(qrow + c * 32 + kh * 8);
    ((v8h*)&qa[c])[1] = *(const v8h*)(qrow + c * 32 + 16 + kh * 8);
  }

  float mq[8];
#pragma unroll
  for (int r = 0; r < 8; r++) mq[r] = mask[b * SEQ + qbase + kh * 8 + r];

  float mrow[8], lrow[8];
#pragma unroll
  for (int r = 0; r < 8; r++) { mrow[r] = -3.0e38f; lrow[r] = 0.f; }
  v8f zero = {};
  v8f ctx[4];
#pragma unroll
  for (int t = 0; t < 4; t++) ctx[t] = zero;

  for (int kb = 0; kb < SEQ; kb += 32) {
    // prefetch next key/value tiles (global_prefetch_b8)
    if (kb + 32 < SEQ) {
      __builtin_prefetch(Kh + (size_t)(b * SEQ + kb + 32 + lm) * HIDDEN + h * HDIM, 0, 0);
      __builtin_prefetch(Vt + ((size_t)bh * HDIM + lm) * SEQ + kb + 32, 0, 0);
    }

    v8f s[2] = {zero, zero};
#pragma unroll
    for (int sub = 0; sub < 2; sub++) {
      const _Float16* krow = Kh + (size_t)(b * SEQ + kb + sub * 16 + lm) * HIDDEN + h * HDIM;
#pragma unroll
      for (int c = 0; c < 2; c++) {
        v16h kf;
        ((v8h*)&kf)[0] = *(const v8h*)(krow + c * 32 + kh * 8);
        ((v8h*)&kf)[1] = *(const v8h*)(krow + c * 32 + 16 + kh * 8);
        s[sub] = __builtin_amdgcn_wmma_f32_16x16x32_f16(
            false, qa[c], false, kf, (short)0, s[sub], false, false);
      }
    }
    float mk0 = mask[b * SEQ + kb + lm];
    float mk1 = mask[b * SEQ + kb + 16 + lm];

    float p0[8], p1[8], alpha[8];
#pragma unroll
    for (int r = 0; r < 8; r++) {
      float e0 = mq[r] * mk0, e1 = mq[r] * mk1;
      float a0 = e0 * (s[0][r] * 0.125f) - (1.f - e0) * 1e9f;
      float a1 = e1 * (s[1][r] * 0.125f) - (1.f - e1) * 1e9f;
      float t = fmaxf(a0, a1);
#pragma unroll
      for (int off = 1; off < 16; off <<= 1) t = fmaxf(t, __shfl_xor(t, off, 32));
      float mnew = fmaxf(mrow[r], t);
      alpha[r] = __expf(mrow[r] - mnew);
      p0[r] = __expf(a0 - mnew);
      p1[r] = __expf(a1 - mnew);
      float ps = p0[r] + p1[r];
#pragma unroll
      for (int off = 1; off < 16; off <<= 1) ps += __shfl_xor(ps, off, 32);
      lrow[r] = lrow[r] * alpha[r] + ps;
      mrow[r] = mnew;
    }

#pragma unroll
    for (int t = 0; t < 4; t++)
#pragma unroll
      for (int r = 0; r < 8; r++) ctx[t][r] *= alpha[r];

    // D-layout -> A-layout transpose of P through LDS
#pragma unroll
    for (int r = 0; r < 8; r++) {
      Pl[(kh * 8 + r) * 40 + lm]      = (_Float16)p0[r];
      Pl[(kh * 8 + r) * 40 + 16 + lm] = (_Float16)p1[r];
    }
    __syncthreads();
    v16h pf;
    ((v8h*)&pf)[0] = *(const v8h*)(&Pl[lm * 40] + kh * 8);
    ((v8h*)&pf)[1] = *(const v8h*)(&Pl[lm * 40] + 16 + kh * 8);
    __syncthreads();

#pragma unroll
    for (int t = 0; t < 4; t++) {
      const _Float16* vr = Vt + ((size_t)bh * HDIM + t * 16 + lm) * SEQ + kb + kh * 8;
      v16h vf;
      ((v8h*)&vf)[0] = *(const v8h*)(vr);
      ((v8h*)&vf)[1] = *(const v8h*)(vr + 16);
      ctx[t] = __builtin_amdgcn_wmma_f32_16x16x32_f16(
          false, pf, false, vf, (short)0, ctx[t], false, false);
    }
  }

#pragma unroll
  for (int r = 0; r < 8; r++) {
    float inv = 1.0f / lrow[r];
    int m = qbase + kh * 8 + r;
#pragma unroll
    for (int t = 0; t < 4; t++) {
      int col = h * HDIM + t * 16 + lm;
      Ctx[(size_t)(b * SEQ + m) * HIDDEN + col] = (_Float16)(ctx[t][r] * inv);
    }
  }
}

// ---------------- launch ----------------
extern "C" void kernel_launch(void* const* d_in, const int* in_sizes, int n_in,
                              void* d_out, int out_size, void* d_ws, size_t ws_size,
                              hipStream_t stream) {
  const float* hidden = (const float*)d_in[0];
  const float* mask   = (const float*)d_in[1];
  const float* Wq = (const float*)d_in[2];
  const float* bq = (const float*)d_in[3];
  const float* Wk = (const float*)d_in[4];
  const float* bk = (const float*)d_in[5];
  const float* Wv = (const float*)d_in[6];
  const float* bv = (const float*)d_in[7];
  const float* Wo = (const float*)d_in[8];
  const float* bo = (const float*)d_in[9];
  float* out = (float*)d_out;

  _Float16* ws = (_Float16*)d_ws;
  size_t o = 0;
  _Float16* Xh  = ws + o; o += (size_t)ROWS * HIDDEN;
  _Float16* Wqt = ws + o; o += (size_t)HIDDEN * HIDDEN;
  _Float16* Wkt = ws + o; o += (size_t)HIDDEN * HIDDEN;
  _Float16* Wvt = ws + o; o += (size_t)HIDDEN * HIDDEN;
  _Float16* Wot = ws + o; o += (size_t)HIDDEN * HIDDEN;
  _Float16* Qh  = ws + o; o += (size_t)ROWS * HIDDEN;
  _Float16* Kh  = ws + o; o += (size_t)ROWS * HIDDEN;
  _Float16* Vt  = ws + o; o += (size_t)ROWS * HIDDEN;   // [48][64][2048]
  _Float16* Ch  = ws + o; o += (size_t)ROWS * HIDDEN;

  {
    int n = ROWS * HIDDEN;
    k_cvt_f16<<<(n + 255) / 256, 256, 0, stream>>>(hidden, Xh, n);
  }
  dim3 tg(HIDDEN / 32, HIDDEN / 32), tb(32, 8);
  k_transposeW<<<tg, tb, 0, stream>>>(Wq, Wqt);
  k_transposeW<<<tg, tb, 0, stream>>>(Wk, Wkt);
  k_transposeW<<<tg, tb, 0, stream>>>(Wv, Wvt);
  k_transposeW<<<tg, tb, 0, stream>>>(Wo, Wot);

  dim3 gg(ROWS / BM, HIDDEN / BN);
  k_gemm<0><<<gg, 256, 0, stream>>>(Xh, Wqt, bq, Qh, nullptr);
  k_gemm<0><<<gg, 256, 0, stream>>>(Xh, Wkt, bk, Kh, nullptr);
  k_gemm<1><<<gg, 256, 0, stream>>>(Xh, Wvt, bv, Vt, nullptr);

  dim3 ag(SEQ / 16, BATCH * NHEADS);
  k_attn<<<ag, 32, 0, stream>>>(Qh, Kh, Vt, mask, Ch);

  k_gemm<2><<<gg, 256, 0, stream>>>(Ch, Wot, bo, nullptr, out);
}
